// coloring_loss_71811853189838
// MI455X (gfx1250) — compile-verified, matrix operand order checked by reading, not run on hardware
//
#include <hip/hip_runtime.h>

typedef __attribute__((ext_vector_type(16))) _Float16 v16h;
typedef __attribute__((ext_vector_type(8)))  float    v8f;
typedef __attribute__((ext_vector_type(4)))  float    v4f;

#define BATCH 8
#define NN 2048
#define EPS 0.01f
#define TILE_COLS 512                       // 32 lanes * 16 elements
#define TILES_PER_ROW (NN / TILE_COLS)      // 4
#define TOTAL_TILES (BATCH * NN * TILES_PER_ROW)  // 65536

__global__ void zero_out_kernel(float* out) {
    if (threadIdx.x == 0 && blockIdx.x == 0) out[0] = 0.0f;
}

__global__ __launch_bounds__(256)
void coloring_mark_kernel(const float* __restrict__ W,
                          const float* __restrict__ pred,
                          float* __restrict__ out) {
    const int lane        = threadIdx.x & 31;
    const int waveInBlock = threadIdx.x >> 5;
    const int wavesPerBlk = blockDim.x >> 5;
    const int wave        = blockIdx.x * wavesPerBlk + waveInBlock;
    const int nWaves      = gridDim.x * wavesPerBlk;

    // B matrix = all ones (f16). Sum(C) = 16 * Sum(A) regardless of A layout,
    // so any bijection of mask elements onto A slots is valid for a global sum.
    v16h ones;
#pragma unroll
    for (int e = 0; e < 16; ++e) ones[e] = (_Float16)1.0f;

    v8f acc = {};

    for (int t = wave; t < TOTAL_TILES; t += nWaves) {   // wave-uniform bound: EXEC stays full
        const int row     = t >> 2;                      // t / TILES_PER_ROW
        const int colBase = (t & 3) * TILE_COLS;
        const int b       = row >> 11;                   // row / NN
        const int i       = row & (NN - 1);              // row % NN

        const float pi = pred[b * NN + i];

        // Channel 1 of W only: plane offset ((b*2 + 1)*NN + i)*NN
        const float* wRow = W + ((size_t)((b * 2 + 1) * NN + i)) * NN
                              + colBase + lane * 16;
        const float* pRow = pred + b * NN + colBase + lane * 16;

        const v4f w0 = *(const v4f*)(wRow + 0);
        const v4f w1 = *(const v4f*)(wRow + 4);
        const v4f w2 = *(const v4f*)(wRow + 8);
        const v4f w3 = *(const v4f*)(wRow + 12);
        const v4f p0 = *(const v4f*)(pRow + 0);
        const v4f p1 = *(const v4f*)(pRow + 4);
        const v4f p2 = *(const v4f*)(pRow + 8);
        const v4f p3 = *(const v4f*)(pRow + 12);

        v16h a;
#pragma unroll
        for (int e = 0; e < 4; ++e) {
            a[e + 0]  = (w0[e] == 1.0f && __builtin_fabsf(pi - p0[e]) < EPS)
                        ? (_Float16)1.0f : (_Float16)0.0f;
            a[e + 4]  = (w1[e] == 1.0f && __builtin_fabsf(pi - p1[e]) < EPS)
                        ? (_Float16)1.0f : (_Float16)0.0f;
            a[e + 8]  = (w2[e] == 1.0f && __builtin_fabsf(pi - p2[e]) < EPS)
                        ? (_Float16)1.0f : (_Float16)0.0f;
            a[e + 12] = (w3[e] == 1.0f && __builtin_fabsf(pi - p3[e]) < EPS)
                        ? (_Float16)1.0f : (_Float16)0.0f;
        }

        // D = A x Ones + C  -> accumulates 16 * (count of set mask bits in tile)
        acc = __builtin_amdgcn_wmma_f32_16x16x32_f16(
            /*neg_a=*/false, a, /*neg_b=*/false, ones,
            /*c_mod=*/(short)0, acc, /*reuse_a=*/false, /*reuse_b=*/false);
    }

    // Per-lane partial: sum the 8 C VGPRs
    float s = acc[0] + acc[1] + acc[2] + acc[3] + acc[4] + acc[5] + acc[6] + acc[7];

    // wave32 butterfly reduction
#pragma unroll
    for (int off = 16; off >= 1; off >>= 1)
        s += __shfl_xor(s, off, 32);

    if (lane == 0)
        atomicAdd(out, s * (1.0f / 16.0f));   // undo the 16x from ones-B; exact in f32
}

extern "C" void kernel_launch(void* const* d_in, const int* in_sizes, int n_in,
                              void* d_out, int out_size, void* d_ws, size_t ws_size,
                              hipStream_t stream) {
    const float* W    = (const float*)d_in[0];   // (B, 2, N, N) f32
    const float* pred = (const float*)d_in[1];   // (B, N) f32
    // d_in[2] (tgt) unused by the loss
    float* out = (float*)d_out;

    zero_out_kernel<<<1, 32, 0, stream>>>(out);

    const int threads = 256;                     // 8 wave32s per block
    const int blocks  = 4096;                    // 32768 waves, 2 tiles each
    coloring_mark_kernel<<<blocks, threads, 0, stream>>>(W, pred, out);
}
